// LocalAttention_20667382628761
// MI455X (gfx1250) — matrix operand order repaired and checked by reading, not compiled, not run on hardware
//
#include <hip/hip_runtime.h>

// LocalAttention (banded, window=128) for MI455X / gfx1250.
// Split-bf16 (compensated, TF32x3-style) WMMA for the big K=256 GEMMs,
// fp32 WMMA 16x16x4 for the post-softmax PV product.
// B=4, S=4096, H=256. Output: [B, H] = mean over S of attention context.

#define BATCH 4
#define SEQ   4096
#define HID   256
#define WIN   128

typedef __attribute__((ext_vector_type(2)))  float  v2f;
typedef __attribute__((ext_vector_type(8)))  float  v8f;
typedef __attribute__((ext_vector_type(4)))  __bf16 v4bf;
typedef __attribute__((ext_vector_type(8)))  __bf16 v8bf;
typedef __attribute__((ext_vector_type(16))) __bf16 v16bf;

__device__ __forceinline__ v8f wmma4(v2f a, v2f b, v8f c) {
  // D = A(16x4 f32) * B(4x16 f32) + C
  return __builtin_amdgcn_wmma_f32_16x16x4_f32(false, a, false, b, (short)0, c,
                                               false, false);
}
__device__ __forceinline__ v8f wmma32(v16bf a, v16bf b, v8f c) {
  // D = A(16x32 bf16) * B(32x16 bf16) + C
  return __builtin_amdgcn_wmma_f32_16x16x32_bf16(false, a, false, b, (short)0, c,
                                                 false, false);
}

__device__ __forceinline__ v8f v8f_zero() {
  v8f z;
#pragma unroll
  for (int i = 0; i < 8; ++i) z[i] = 0.0f;
  return z;
}

// A-fragment (16x32 bf16): lanes 0-15 hold K = {hi8..hi8+7, 16+hi8..}, per ISA
__device__ __forceinline__ v16bf load_a_frag(const __bf16* row, int hi) {
  v8bf c0 = *(const v8bf*)(row + hi * 8);
  v8bf c1 = *(const v8bf*)(row + 16 + hi * 8);
  return __builtin_shufflevector(c0, c1, 0, 1, 2, 3, 4, 5, 6, 7,
                                 8, 9, 10, 11, 12, 13, 14, 15);
}
// B-fragment (32x16 bf16): lanes 0-15 hold K = {hi8..hi8+7, 16+hi8..}, same as A
__device__ __forceinline__ v16bf load_b_frag(const __bf16* row, int hi) {
  v8bf c0 = *(const v8bf*)(row + hi * 8);
  v8bf c1 = *(const v8bf*)(row + 16 + hi * 8);
  return __builtin_shufflevector(c0, c1, 0, 1, 2, 3, 4, 5, 6, 7,
                                 8, 9, 10, 11, 12, 13, 14, 15);
}

__device__ __forceinline__ void bf16_split(float f, __bf16& h, __bf16& l) {
  h = (__bf16)f;
  l = (__bf16)(f - (float)h);
}

// ---------------------------------------------------------------------------
// Pass 0: zero the output (harness poisons d_out with 0xAA).
// ---------------------------------------------------------------------------
__global__ void la_zero_out(float* __restrict__ out, int n) {
  int i = blockIdx.x * blockDim.x + threadIdx.x;
  if (i < n) out[i] = 0.0f;
}

// ---------------------------------------------------------------------------
// Pass 1: QKV projection, split-bf16. X[16384x256] * W[256x256] + b -> ws.
// Block: 128x128 output tile; 8 waves, each 2x4 16x16 subtiles; K chunk = 32.
// ---------------------------------------------------------------------------
#define P1_TM 128
#define P1_TN 128
#define P1_KC 32
#define XB_STRIDE 40  // 32 + pad8: keeps b128 frags 16B-aligned, banks +20dw/row

__global__ __launch_bounds__(256) void la_qkv(
    const float* __restrict__ x,
    const float* __restrict__ Wq, const float* __restrict__ bq,
    const float* __restrict__ Wk, const float* __restrict__ bk,
    const float* __restrict__ Wv, const float* __restrict__ bv,
    float* __restrict__ qo, float* __restrict__ ko, float* __restrict__ vo) {
  __shared__ __bf16 Xh[P1_TM * XB_STRIDE];   // X tile, bf16 high parts
  __shared__ __bf16 Xl[P1_TM * XB_STRIDE];   // X tile, bf16 residuals
  __shared__ __bf16 Wth[P1_TN * XB_STRIDE];  // W tile transposed [n][k], high
  __shared__ __bf16 Wtl[P1_TN * XB_STRIDE];  // W tile transposed [n][k], low

  const float* Wm;
  const float* bm;
  float* om;
  if (blockIdx.z == 0)      { Wm = Wq; bm = bq; om = qo; }
  else if (blockIdx.z == 1) { Wm = Wk; bm = bk; om = ko; }
  else                      { Wm = Wv; bm = bv; om = vo; }

  const int m0 = blockIdx.x * P1_TM;
  const int n0 = blockIdx.y * P1_TN;
  const int tid  = threadIdx.x;
  const int wave = tid >> 5;
  const int lane = tid & 31;
  const int hi   = lane >> 4;
  const int l15  = lane & 15;
  const int wm   = wave >> 1;  // 4 M-groups of 32 rows
  const int wn   = wave & 1;   // 2 N-groups of 64 cols

  v8f acc[2][4];
#pragma unroll
  for (int i = 0; i < 2; ++i)
#pragma unroll
    for (int j = 0; j < 4; ++j) acc[i][j] = v8f_zero();

  for (int k0 = 0; k0 < HID; k0 += P1_KC) {
    // stage X tile: 128x32 floats -> hi/lo bf16 (1024 float4, 4/thread)
#pragma unroll
    for (int i = 0; i < 4; ++i) {
      int lin = tid + i * 256;
      int row = lin >> 3;
      int c4  = lin & 7;
      float4 t = *(const float4*)&x[(size_t)(m0 + row) * HID + k0 + c4 * 4];
      v4bf h, l;
#pragma unroll
      for (int e = 0; e < 4; ++e) {
        __bf16 hb, lb;
        bf16_split(((const float*)&t)[e], hb, lb);
        h[e] = hb; l[e] = lb;
      }
      *(v4bf*)&Xh[row * XB_STRIDE + c4 * 4] = h;
      *(v4bf*)&Xl[row * XB_STRIDE + c4 * 4] = l;
    }
    // stage W tile transposed: W[k][n] -> Wt[n][k], hi/lo bf16
#pragma unroll
    for (int i = 0; i < 4; ++i) {
      int lin = tid + i * 256;
      int row = lin >> 5;   // k index 0..31
      int c4  = lin & 31;   // n group of 4
      float4 t = *(const float4*)&Wm[(size_t)(k0 + row) * HID + n0 + c4 * 4];
#pragma unroll
      for (int e = 0; e < 4; ++e) {
        __bf16 hb, lb;
        bf16_split(((const float*)&t)[e], hb, lb);
        int n = c4 * 4 + e;
        Wth[n * XB_STRIDE + row] = hb;
        Wtl[n * XB_STRIDE + row] = lb;
      }
    }
    __syncthreads();

    v16bf ah[2], al[2];
#pragma unroll
    for (int i = 0; i < 2; ++i) {
      const __bf16* rp = &Xh[(wm * 32 + i * 16 + l15) * XB_STRIDE];
      const __bf16* rq = &Xl[(wm * 32 + i * 16 + l15) * XB_STRIDE];
      ah[i] = load_a_frag(rp, hi);
      al[i] = load_a_frag(rq, hi);
    }
#pragma unroll
    for (int j = 0; j < 4; ++j) {
      int n = wn * 64 + j * 16 + l15;
      v16bf bh = load_b_frag(&Wth[n * XB_STRIDE], hi);
      v16bf bl = load_b_frag(&Wtl[n * XB_STRIDE], hi);
#pragma unroll
      for (int i = 0; i < 2; ++i) {
        // compensated product: Ah*Bh + Ah*Bl + Al*Bh  (~fp32 accuracy)
        acc[i][j] = wmma32(al[i], bh, acc[i][j]);
        acc[i][j] = wmma32(ah[i], bl, acc[i][j]);
        acc[i][j] = wmma32(ah[i], bh, acc[i][j]);
      }
    }
    __syncthreads();
  }

  // bias + store (C layout: vgpr r, lane -> row = r + 8*hi, col = l15)
#pragma unroll
  for (int j = 0; j < 4; ++j) {
    int ng = n0 + wn * 64 + j * 16 + l15;
    float bias = bm[ng];
#pragma unroll
    for (int i = 0; i < 2; ++i) {
#pragma unroll
      for (int r = 0; r < 8; ++r) {
        int row = m0 + wm * 32 + i * 16 + r + 8 * hi;
        om[(size_t)row * HID + ng] = acc[i][j][r] + bias;
      }
    }
  }
}

// ---------------------------------------------------------------------------
// Pass 2: banded flash attention + mean. Block: 64 queries (4 waves x 16).
// QK^T: split-bf16 WMMA (K=256).  PV: fp32 WMMA (keeps P exact).
// Dynamic LDS: Qh|Ql 64x264 bf16, Kh|Kl 16x264 bf16, Vs 16x260 f32, Ps f32.
// ---------------------------------------------------------------------------
#define P2_TQ 64
#define QB_STRIDE 264   // bf16 stride: 528B/row (16B-aligned frags, banks +4/row)
#define VS_STRIDE 260   // f32 stride: 1040B/row
#define PS_STRIDE 18

__global__ __launch_bounds__(128) void la_attn(
    const float* __restrict__ qg, const float* __restrict__ kg,
    const float* __restrict__ vg, float* __restrict__ out) {
  extern __shared__ char smem[];
  __bf16* Qh = (__bf16*)smem;                       // 64*264
  __bf16* Ql = Qh + P2_TQ * QB_STRIDE;              // 64*264
  __bf16* Kh = Ql + P2_TQ * QB_STRIDE;              // 16*264
  __bf16* Kl = Kh + 16 * QB_STRIDE;                 // 16*264
  float*  Vs = (float*)(Kl + 16 * QB_STRIDE);       // 16*260
  float*  Ps = Vs + 16 * VS_STRIDE;                 // 4*16*18

  const int b  = blockIdx.y;
  const int q0 = blockIdx.x * P2_TQ;
  const int tid  = threadIdx.x;
  const int wave = tid >> 5;
  const int lane = tid & 31;
  const int hi   = lane >> 4;
  const int l15  = lane & 15;
  const int qb   = q0 + wave * 16;

  // stage Q tile rows 0-15: 16x256 f32 -> hi/lo bf16 (1024 float4, 8/thread)
#pragma unroll
  for (int i = 0; i < 8; ++i) {
    int lin = tid + i * 128;
    int row = lin >> 6;
    int c4  = lin & 63;
    float4 t = *(const float4*)&qg[(size_t)(b * SEQ + q0 + row) * HID + c4 * 4];
    v4bf h, l;
#pragma unroll
    for (int e = 0; e < 4; ++e) {
      __bf16 hb, lb;
      bf16_split(((const float*)&t)[e], hb, lb);
      h[e] = hb; l[e] = lb;
    }
    *(v4bf*)&Qh[row * QB_STRIDE + c4 * 4] = h;
    *(v4bf*)&Ql[row * QB_STRIDE + c4 * 4] = l;
  }
  __syncthreads();

  float mrun[8], lrun[8];
  v8f o[16];
#pragma unroll
  for (int r = 0; r < 8; ++r) { mrun[r] = -1e30f; lrun[r] = 0.0f; }
#pragma unroll
  for (int f = 0; f < 16; ++f) o[f] = v8f_zero();

  const int ktlo = (q0 - WIN > 0 ? q0 - WIN : 0) >> 4;
  const int kthi = ((q0 + P2_TQ + WIN) < SEQ ? (q0 + P2_TQ + WIN) : SEQ) >> 4;

  for (int kt = ktlo; kt < kthi; ++kt) {
    // stage K (hi/lo bf16) and V (f32) tiles: 16x256 each
#pragma unroll
    for (int i = 0; i < 8; ++i) {
      int lin = tid + i * 128;
      int row = lin >> 6;
      int c4  = lin & 63;
      size_t g = (size_t)(b * SEQ + kt * 16 + row) * HID + c4 * 4;
      float4 t = *(const float4*)&kg[g];
      v4bf h, l;
#pragma unroll
      for (int e = 0; e < 4; ++e) {
        __bf16 hb, lb;
        bf16_split(((const float*)&t)[e], hb, lb);
        h[e] = hb; l[e] = lb;
      }
      *(v4bf*)&Kh[row * QB_STRIDE + c4 * 4] = h;
      *(v4bf*)&Kl[row * QB_STRIDE + c4 * 4] = l;
      *(float4*)&Vs[row * VS_STRIDE + c4 * 4] = *(const float4*)&vg[g];
    }
    __syncthreads();

    bool active = !((kt * 16 + 15 < qb - WIN) || (kt * 16 > qb + 15 + WIN));
    if (active) {
      // S = Q * K^T, split-bf16, K=256 in steps of 32 (3 WMMAs per step)
      v8f s = v8f_zero();
#pragma unroll
      for (int kk0 = 0; kk0 < HID; kk0 += 32) {
        v16bf a_h = load_a_frag(&Qh[(wave * 16 + l15) * QB_STRIDE + kk0], hi);
        v16bf a_l = load_a_frag(&Ql[(wave * 16 + l15) * QB_STRIDE + kk0], hi);
        v16bf b_h = load_b_frag(&Kh[l15 * QB_STRIDE + kk0], hi);
        v16bf b_l = load_b_frag(&Kl[l15 * QB_STRIDE + kk0], hi);
        s = wmma32(a_l, b_h, s);
        s = wmma32(a_h, b_l, s);
        s = wmma32(a_h, b_h, s);
      }

      // online softmax; row = r + 8*hi, col (key) = l15
      const int kj = kt * 16 + l15;
      float rowscale[8];
#pragma unroll
      for (int r = 0; r < 8; ++r) {
        float val = s[r] * 0.0625f;  // 1/sqrt(256)
        int qi = qb + r + 8 * hi;
        bool ok = (kj >= qi - WIN) && (kj <= qi + WIN);
        val = ok ? val : -1e30f;
        float tm = val;
#pragma unroll
        for (int msk = 1; msk <= 8; msk <<= 1) tm = fmaxf(tm, __shfl_xor(tm, msk, 32));
        float mn = fmaxf(mrun[r], tm);
        float sc = __expf(mrun[r] - mn);
        float p  = __expf(val - mn);
        float ts = p;
#pragma unroll
        for (int msk = 1; msk <= 8; msk <<= 1) ts += __shfl_xor(ts, msk, 32);
        lrun[r] = lrun[r] * sc + ts;
        mrun[r] = mn;
        rowscale[r] = sc;
        Ps[(wave * 16 + r + 8 * hi) * PS_STRIDE + l15] = p;  // P -> A layout
      }
#pragma unroll
      for (int f = 0; f < 16; ++f)
#pragma unroll
        for (int r = 0; r < 8; ++r) o[f][r] *= rowscale[r];

      asm volatile("s_wait_dscnt 0" ::: "memory");

      // O += P * V  (fp32 WMMA: 16 feature subtiles, K=16 in steps of 4)
#pragma unroll
      for (int f = 0; f < 16; ++f) {
#pragma unroll
        for (int kk2 = 0; kk2 < 4; ++kk2) {
          v2f a = *(const v2f*)&Ps[(wave * 16 + l15) * PS_STRIDE + kk2 * 4 + 2 * hi];
          v2f bb;
          bb.x = Vs[(kk2 * 4 + 2 * hi) * VS_STRIDE + f * 16 + l15];
          bb.y = Vs[(kk2 * 4 + 2 * hi + 1) * VS_STRIDE + f * 16 + l15];
          o[f] = wmma4(a, bb, o[f]);
        }
      }
    }
    __syncthreads();
  }

  // normalize, reduce 16 rows, scale by 1/S, accumulate into out[b][:]
  float invl[8];
#pragma unroll
  for (int r = 0; r < 8; ++r) invl[r] = 1.0f / lrun[r];
#pragma unroll
  for (int f = 0; f < 16; ++f) {
    float ssum = 0.0f;
#pragma unroll
    for (int r = 0; r < 8; ++r) ssum += o[f][r] * invl[r];
    ssum += __shfl_xor(ssum, 16, 32);
    if (hi == 0)
      atomicAdd(&out[b * HID + f * 16 + l15], ssum * (1.0f / (float)SEQ));
  }
}

// ---------------------------------------------------------------------------
extern "C" void kernel_launch(void* const* d_in, const int* in_sizes, int n_in,
                              void* d_out, int out_size, void* d_ws, size_t ws_size,
                              hipStream_t stream) {
  const float* x  = (const float*)d_in[0];
  const float* Wq = (const float*)d_in[1];
  const float* bq = (const float*)d_in[2];
  const float* Wk = (const float*)d_in[3];
  const float* bk = (const float*)d_in[4];
  const float* Wv = (const float*)d_in[5];
  const float* bv = (const float*)d_in[6];
  // d_in[7] = window_size (==128, baked into WIN)

  float* out = (float*)d_out;
  float* qws = (float*)d_ws;
  float* kws = qws + (size_t)BATCH * SEQ * HID;
  float* vws = kws + (size_t)BATCH * SEQ * HID;

  la_zero_out<<<(out_size + 255) / 256, 256, 0, stream>>>(out, out_size);

  la_qkv<<<dim3(BATCH * SEQ / P1_TM, HID / P1_TN, 3), 256, 0, stream>>>(
      x, Wq, bq, Wk, bk, Wv, bv, qws, kws, vws);

  size_t attn_lds = (size_t)(P2_TQ + 16) * QB_STRIDE * 2 * sizeof(__bf16)  // Qh/Ql/Kh/Kl
                  + (size_t)16 * VS_STRIDE * sizeof(float)                 // Vs
                  + (size_t)4 * 16 * PS_STRIDE * sizeof(float);            // Ps
  la_attn<<<dim3(SEQ / P2_TQ, BATCH), 128, attn_lds, stream>>>(qws, kws, vws, out);
}
